// ElemAttn_24197845746072
// MI455X (gfx1250) — matrix operand order repaired and checked by reading, not compiled, not run on hardware
//
#include <hip/hip_runtime.h>

typedef __attribute__((ext_vector_type(16))) __bf16 v16bf;
typedef __attribute__((ext_vector_type(8)))  __bf16 v8bf;
typedef __attribute__((ext_vector_type(8)))  float  v8f;

#define N_ATOMS 4096
#define F_DIM   256
#define H_HEADS 8
#define D_HEAD  32
#define G_GRP   128

__device__ __forceinline__ float silu_f(float s) { return s / (1.0f + __expf(-s)); }

__device__ __forceinline__ void atomic_fadd_agent(float* p, float v) {
    __hip_atomic_fetch_add(p, v, __ATOMIC_RELAXED, __HIP_MEMORY_SCOPE_AGENT);
}

// ---------------- conversion / init helpers ----------------

__global__ void k_convert_bf16(const float* __restrict__ src, __bf16* __restrict__ dst, int n) {
    int i = blockIdx.x * blockDim.x + threadIdx.x;
    if (i < n) dst[i] = (__bf16)src[i];
}

__global__ void k_zero_f32(float* __restrict__ p, int n) {
    int i = blockIdx.x * blockDim.x + threadIdx.x;
    if (i < n) p[i] = 0.0f;
}

// grp_start[g] = lower_bound(seg, g); grp_start[G] = N  (seg is sorted)
__global__ void k_grp_start(const int* __restrict__ seg, int* __restrict__ gs) {
    int g = blockIdx.x * blockDim.x + threadIdx.x;
    if (g > G_GRP) return;
    int lo = 0, hi = N_ATOMS;
    while (lo < hi) {
        int mid = (lo + hi) >> 1;
        if (seg[mid] < g) lo = mid + 1; else hi = mid;
    }
    gs[g] = lo;
}

// ---------------- fused QKV projection (bf16 WMMA) ----------------
// Each wave computes a 16x64 output strip of one projection: 4 accumulators
// share one A (x) tile per K-strip -> 4x less A traffic, 4 WMMAs per strip.
// q,k row-major bf16 [N][F]; v stored transposed [F][N].
__global__ __launch_bounds__(32) void k_qkv(
    const __bf16* __restrict__ xb,
    const __bf16* __restrict__ wq, const __bf16* __restrict__ wk, const __bf16* __restrict__ wv,
    const float*  __restrict__ bq, const float*  __restrict__ bk, const float*  __restrict__ bv,
    __bf16* __restrict__ qb, __bf16* __restrict__ kb, __bf16* __restrict__ vT)
{
    const int L   = threadIdx.x;
    const int lhi = (L >= 16) ? 1 : 0;
    const int lm  = L & 15;

    int bid = blockIdx.x;
    int p   = bid % 3;                          // 0=q 1=k 2=v
    int f4  = (bid / 3) % (F_DIM / 64);
    int mt  = bid / (3 * (F_DIM / 64));
    int m0  = mt * 16, f0 = f4 * 64;

    const __bf16* W    = (p == 0) ? wq : (p == 1) ? wk : wv;
    const float*  bias = (p == 0) ? bq : (p == 1) ? bk : bv;

    const __bf16* arow = xb + (m0 + lm) * F_DIM;

    v8f acc[4] = {v8f{}, v8f{}, v8f{}, v8f{}};
#pragma unroll
    for (int kk = 0; kk < F_DIM; kk += 32) {
        v16bf A;
        const __bf16* ap = arow + kk + (lhi ? 8 : 0);        // A-layout: K 0-7/16-23 (lo) | 8-15/24-31 (hi)
        ((v8bf*)&A)[0] = *(const v8bf*)(ap);
        ((v8bf*)&A)[1] = *(const v8bf*)(ap + 16);
#pragma unroll
        for (int j = 0; j < 4; ++j) {
            v16bf B;                                          // B-layout: K 0-7/16-23 (lo) | 8-15/24-31 (hi), N = lm
            const __bf16* bp = W + (f0 + j * 16 + lm) * F_DIM + kk + (lhi ? 8 : 0);
            ((v8bf*)&B)[0] = *(const v8bf*)(bp);
            ((v8bf*)&B)[1] = *(const v8bf*)(bp + 16);
            acc[j] = __builtin_amdgcn_wmma_f32_16x16x32_bf16(false, A, false, B, (short)0, acc[j], false, false);
        }
    }

#pragma unroll
    for (int j = 0; j < 4; ++j) {
        int f = f0 + j * 16 + lm;
        float bval = bias[f];
        if (p < 2) {
            __bf16* out = (p == 0) ? qb : kb;
#pragma unroll
            for (int r = 0; r < 8; ++r) {                     // C-layout: M = r (+8 hi lanes), N = lm
                int m = m0 + r + (lhi ? 8 : 0);
                out[m * F_DIM + f] = (__bf16)(acc[j][r] + bval);
            }
        } else {
#pragma unroll
            for (int r = 0; r < 8; ++r) {                     // v transposed: vT[f][n]
                int m = m0 + r + (lhi ? 8 : 0);
                vT[f * N_ATOMS + m] = (__bf16)(acc[j][r] + bval);
            }
        }
    }
}

// ---------------- block-diagonal attention (bf16 WMMA) ----------------
// One wave per (head, 16-row tile). Iterates only over the segment column range.
__global__ __launch_bounds__(32) void k_attn(
    const __bf16* __restrict__ qb, const __bf16* __restrict__ kb, const __bf16* __restrict__ vT,
    const int* __restrict__ seg, const int* __restrict__ gs,
    float* __restrict__ Ssum)
{
    __shared__ __bf16 sS[16 * 32];   // masked-silu S tile, 16 rows x 32 cols
    const int L   = threadIdx.x;
    const int lhi = (L >= 16) ? 1 : 0;
    const int lm  = L & 15;

    int h  = blockIdx.x >> 8;        // N/16 = 256 row tiles
    int mt = blockIdx.x & 255;
    int m0 = mt * 16;

    // A (q) tile: 16x32 bf16 in A-layout
    v16bf Aq;
    const __bf16* qrow = qb + (m0 + lm) * F_DIM + h * D_HEAD + (lhi ? 8 : 0);
    ((v8bf*)&Aq)[0] = *(const v8bf*)(qrow);
    ((v8bf*)&Aq)[1] = *(const v8bf*)(qrow + 16);

    int sr[8];                       // segment id per C-layout row
#pragma unroll
    for (int r = 0; r < 8; ++r) sr[r] = seg[m0 + r + (lhi ? 8 : 0)];

    int segFirst = seg[m0];
    int segLast  = seg[m0 + 15];
    int cstart   = gs[segFirst] & ~31;     // strip-aligned; out-of-segment cols are masked anyway
    int cend     = gs[segLast + 1];

    v8f acc0 = {}, acc1 = {};

    for (int c0 = cstart; c0 < cend; c0 += 32) {
        // ---- S = silu(q k^T) * mask for two 16-col subtiles, staged to LDS ----
#pragma unroll
        for (int sub = 0; sub < 2; ++sub) {
            int c    = c0 + sub * 16;
            int col  = c + lm;
            int cseg = (col < N_ATOMS) ? seg[col] : -1;
            int ca   = (col < N_ATOMS) ? col : (N_ATOMS - 1);

            v16bf Bk;  // B-layout: column = attention col, K = d (0-7/16-23 lo / 8-15/24-31 hi lanes)
            const __bf16* krow = kb + ca * F_DIM + h * D_HEAD + (lhi ? 8 : 0);
            ((v8bf*)&Bk)[0] = *(const v8bf*)(krow);
            ((v8bf*)&Bk)[1] = *(const v8bf*)(krow + 16);

            v8f S = {};
            S = __builtin_amdgcn_wmma_f32_16x16x32_bf16(false, Aq, false, Bk, (short)0, S, false, false);
#pragma unroll
            for (int r = 0; r < 8; ++r) {
                float s = silu_f(S[r]);
                s = (sr[r] == cseg) ? s : 0.0f;
                sS[(r + (lhi ? 8 : 0)) * 32 + sub * 16 + lm] = (__bf16)s;
            }
        }
        __syncthreads();   // single-wave workgroup: just the LDS wait

        // ---- reload S as A-matrix (16x32), multiply by V (32x32 -> two 16-col WMMAs) ----
        v16bf A2;
        const __bf16* srow = sS + lm * 32 + (lhi ? 8 : 0);
        ((v8bf*)&A2)[0] = *(const v8bf*)(srow);
        ((v8bf*)&A2)[1] = *(const v8bf*)(srow + 16);
        {
            v16bf Bv;  // vT[f][n]: contiguous in n => two b128 loads
            const __bf16* vrow = vT + (h * D_HEAD + lm) * N_ATOMS + c0 + (lhi ? 8 : 0);
            ((v8bf*)&Bv)[0] = *(const v8bf*)(vrow);
            ((v8bf*)&Bv)[1] = *(const v8bf*)(vrow + 16);
            acc0 = __builtin_amdgcn_wmma_f32_16x16x32_bf16(false, A2, false, Bv, (short)0, acc0, false, false);
        }
        {
            v16bf Bv;
            const __bf16* vrow = vT + (h * D_HEAD + 16 + lm) * N_ATOMS + c0 + (lhi ? 8 : 0);
            ((v8bf*)&Bv)[0] = *(const v8bf*)(vrow);
            ((v8bf*)&Bv)[1] = *(const v8bf*)(vrow + 16);
            acc1 = __builtin_amdgcn_wmma_f32_16x16x32_bf16(false, A2, false, Bv, (short)0, acc1, false, false);
        }
        __syncthreads();   // protect LDS reuse next strip
    }

    // ---- fused segment-sum: scatter rows into per-group accumulator ----
#pragma unroll
    for (int r = 0; r < 8; ++r) {
        float* dst = Ssum + sr[r] * F_DIM + h * D_HEAD;
        atomic_fadd_agent(dst + lm,      acc0[r]);
        atomic_fadd_agent(dst + 16 + lm, acc1[r]);
    }
}

// ---------------- tiny output GEMMs (fp32, 128x256x256 each) ----------------
__global__ void k_o1(const float* __restrict__ Ssum, const float* __restrict__ Wo1,
                     const float* __restrict__ bo1, const int* __restrict__ gs,
                     float* __restrict__ z1)
{
    int t = blockIdx.x * blockDim.x + threadIdx.x;
    if (t >= G_GRP * F_DIM) return;
    int g = t / F_DIM, f = t % F_DIM;
    const float* a = Ssum + g * F_DIM;
    const float* w = Wo1 + f * F_DIM;
    float s = 0.0f;
#pragma unroll 8
    for (int c = 0; c < F_DIM; ++c) s += a[c] * w[c];
    float cnt = (float)(gs[g + 1] - gs[g]);   // segment_sum of the bias term
    z1[t] = s + cnt * bo1[f];
}

__global__ void k_o2(const float* __restrict__ z1, const float* __restrict__ Wo2,
                     const float* __restrict__ bo2, float* __restrict__ out)
{
    int t = blockIdx.x * blockDim.x + threadIdx.x;
    if (t >= G_GRP * F_DIM) return;
    int g = t / F_DIM, f = t % F_DIM;
    const float* a = z1 + g * F_DIM;
    const float* w = Wo2 + f * F_DIM;
    float s = 0.0f;
#pragma unroll 8
    for (int c = 0; c < F_DIM; ++c) s += a[c] * w[c];
    out[t] = silu_f(s + bo2[f]);
}

// ---------------- host launch ----------------
extern "C" void kernel_launch(void* const* d_in, const int* in_sizes, int n_in,
                              void* d_out, int out_size, void* d_ws, size_t ws_size,
                              hipStream_t stream)
{
    (void)in_sizes; (void)n_in; (void)out_size; (void)ws_size;
    const float* x    = (const float*)d_in[0];
    const int*   seg  = (const int*)d_in[1] + N_ATOMS;   // elem_index row 1
    const float* Wq   = (const float*)d_in[2];
    const float* bq   = (const float*)d_in[3];
    const float* Wk   = (const float*)d_in[4];
    const float* bk   = (const float*)d_in[5];
    const float* Wv   = (const float*)d_in[6];
    const float* bv   = (const float*)d_in[7];
    const float* Wo1  = (const float*)d_in[8];
    const float* bo1  = (const float*)d_in[9];
    const float* Wo2  = (const float*)d_in[10];
    const float* bo2  = (const float*)d_in[11];
    float* out = (float*)d_out;

    char* ws = (char*)d_ws;
    size_t off = 0;
    __bf16* xb  = (__bf16*)(ws + off); off += (size_t)N_ATOMS * F_DIM * 2;
    __bf16* wqb = (__bf16*)(ws + off); off += (size_t)F_DIM * F_DIM * 2;
    __bf16* wkb = (__bf16*)(ws + off); off += (size_t)F_DIM * F_DIM * 2;
    __bf16* wvb = (__bf16*)(ws + off); off += (size_t)F_DIM * F_DIM * 2;
    __bf16* qb  = (__bf16*)(ws + off); off += (size_t)N_ATOMS * F_DIM * 2;
    __bf16* kb  = (__bf16*)(ws + off); off += (size_t)N_ATOMS * F_DIM * 2;
    __bf16* vT  = (__bf16*)(ws + off); off += (size_t)N_ATOMS * F_DIM * 2;
    int*    gs  = (int*)(ws + off);    off += 1024;
    float*  Ssum= (float*)(ws + off);  off += (size_t)G_GRP * F_DIM * 4;
    float*  z1  = (float*)(ws + off);  off += (size_t)G_GRP * F_DIM * 4;

    int nx = N_ATOMS * F_DIM, nw = F_DIM * F_DIM, ng = G_GRP * F_DIM;
    k_convert_bf16<<<(nx + 255) / 256, 256, 0, stream>>>(x,  xb,  nx);
    k_convert_bf16<<<(nw + 255) / 256, 256, 0, stream>>>(Wq, wqb, nw);
    k_convert_bf16<<<(nw + 255) / 256, 256, 0, stream>>>(Wk, wkb, nw);
    k_convert_bf16<<<(nw + 255) / 256, 256, 0, stream>>>(Wv, wvb, nw);
    k_zero_f32<<<(ng + 255) / 256, 256, 0, stream>>>(Ssum, ng);
    k_grp_start<<<1, 256, 0, stream>>>(seg, gs);

    k_qkv<<<(N_ATOMS / 16) * (F_DIM / 64) * 3, 32, 0, stream>>>(
        xb, wqb, wkb, wvb, bq, bk, bv, qb, kb, vT);

    k_attn<<<H_HEADS * (N_ATOMS / 16), 32, 0, stream>>>(qb, kb, vT, seg, gs, Ssum);

    k_o1<<<(ng + 255) / 256, 256, 0, stream>>>(Ssum, Wo1, bo1, gs, z1);
    k_o2<<<(ng + 255) / 256, 256, 0, stream>>>(z1, Wo2, bo2, out);
}
